// DeepGG_68908455297284
// MI455X (gfx1250) — matrix operand order repaired and checked by reading, not compiled, not run on hardware
//
#include <hip/hip_runtime.h>
#include <math.h>

typedef __attribute__((ext_vector_type(16))) _Float16 v16h;
typedef __attribute__((ext_vector_type(8)))  float    v8f;

// ================= WMMA fragment helpers (gfx1250, wave32) =================
// Packed operand buffers hold fragments in exact per-lane register order:
//   A: [rowTile*KT + kt][lane][16 halves]   (ISA 7.12.2 16-bit A layout)
//   B: [kt*CT + ct][lane][16 halves]        (ISA 16-bit B layout)
// so each v16h is one 32-byte contiguous load per lane (2x global_load_b128).

__device__ __forceinline__ v16h load_pa(const _Float16* __restrict__ P, int wid, int KT,
                                        int kt, int lane) {
  return *(const v16h*)(P + (((size_t)wid * KT + kt) * 32 + lane) * 16);
}
__device__ __forceinline__ v16h load_pb(const _Float16* __restrict__ P, int kt, int CT,
                                        int ct, int lane) {
  return *(const v16h*)(P + (((size_t)kt * CT + ct) * 32 + lane) * 16);
}

__device__ __forceinline__ v8f wmma16(v16h a, v16h b, v8f c) {
  return __builtin_amdgcn_wmma_f32_16x16x32_f16(false, a, false, b, (short)0, c, false, false);
}

__device__ __forceinline__ v8f zero8() {
  v8f z;
#pragma unroll
  for (int i = 0; i < 8; ++i) z[i] = 0.0f;
  return z;
}

__device__ __forceinline__ float sigf(float x) { return 1.0f / (1.0f + expf(-x)); }

// ================= pack kernels (fp32 row-major -> f16 fragment order) =================
// A fragments: row = rt*16 + (lane&15); khi = (lane>>4)*8;
// element i: v=i/2, odd=i&1, kk = (v<4 ? 2v : 16+2(v-4)) + odd; K = kt*32 + khi + kk.
__global__ void k_pack_a(const float* __restrict__ A, int lda, int KT, int nRowTiles,
                         _Float16* __restrict__ out) {
  size_t total = (size_t)nRowTiles * KT * 512;
  size_t tid = (size_t)blockIdx.x * blockDim.x + threadIdx.x;
  size_t stride = (size_t)gridDim.x * blockDim.x;
  for (; tid < total; tid += stride) {
    int i = (int)(tid & 15);
    int lane = (int)((tid >> 4) & 31);
    size_t tIdx = tid >> 9;
    int kt = (int)(tIdx % KT);
    size_t rt = tIdx / KT;
    int row = (int)(rt * 16 + (lane & 15));
    int khi = (lane >> 4) << 3;
    int v = i >> 1, odd = i & 1;
    int kk = ((v < 4) ? (v << 1) : (16 + ((v - 4) << 1))) + odd;
    out[tid] = (_Float16)A[(size_t)row * lda + kt * 32 + khi + kk];
  }
}

// B fragments: col = ct*16 + (lane&15); K = kt*32 + (lane>>4)*8 + (i&7) + (i>>3)*16.
__global__ void k_pack_b(const float* __restrict__ B, int ldb, int KTt, int CTt,
                         _Float16* __restrict__ out) {
  size_t total = (size_t)KTt * CTt * 512;
  size_t tid = (size_t)blockIdx.x * blockDim.x + threadIdx.x;
  size_t stride = (size_t)gridDim.x * blockDim.x;
  for (; tid < total; tid += stride) {
    int i = (int)(tid & 15);
    int lane = (int)((tid >> 4) & 31);
    size_t tIdx = tid >> 9;
    int ct = (int)(tIdx % CTt);
    int kt = (int)(tIdx / CTt);
    int col = ct * 16 + (lane & 15);
    int kAbs = kt * 32 + ((lane >> 4) << 3) + (i & 7) + ((i >> 3) << 4);
    out[tid] = (_Float16)B[(size_t)kAbs * ldb + col];
  }
}

// ================= utility kernels =================
__global__ void k_zero(float* p, size_t n) {
  size_t i = (size_t)blockIdx.x * blockDim.x + threadIdx.x;
  size_t s = (size_t)gridDim.x * blockDim.x;
  for (; i < n; i += s) p[i] = 0.0f;
}

__global__ void k_deg(const int* __restrict__ src, const int* __restrict__ dst,
                      float* indeg, float* outdeg, int E) {
  int e = blockIdx.x * blockDim.x + threadIdx.x;
  if (e >= E) return;
  atomicAdd(&indeg[dst[e]], 1.0f);
  atomicAdd(&outdeg[src[e]], 1.0f);
}

// ============ msg_src = hv @ W_src  (N,64)x(64,128), packed operands ============
__global__ __launch_bounds__(256) void k_msgsrc(const _Float16* __restrict__ APack,
                                                const _Float16* __restrict__ BPack,
                                                float* __restrict__ out, int nTiles) {
  int lane = threadIdx.x & 31;
  int wid  = blockIdx.x * (blockDim.x >> 5) + (threadIdx.x >> 5);
  if (wid >= nTiles) return;  // wave-uniform exit; EXEC all-ones for WMMA
  int row0 = wid << 4;
  v8f c[8];
#pragma unroll
  for (int i = 0; i < 8; ++i) c[i] = zero8();
#pragma unroll
  for (int kt = 0; kt < 2; ++kt) {
    v16h a = load_pa(APack, wid, 2, kt, lane);
#pragma unroll
    for (int ct = 0; ct < 8; ++ct)
      c[ct] = wmma16(a, load_pb(BPack, kt, 8, ct, lane), c[ct]);
  }
  int mB = (lane >> 4) << 3, cl = lane & 15;
#pragma unroll
  for (int ct = 0; ct < 8; ++ct)
#pragma unroll
    for (int v = 0; v < 8; ++v)
      out[(size_t)(row0 + mB + v) * 128 + ct * 16 + cl] = c[ct][v];
}

// ==== fused edge GEMM + gather + scatter: agg[dst] += hePack@W_edge + msg_src[src] ====
__global__ __launch_bounds__(256) void k_edge(const _Float16* __restrict__ APack,
                                              const _Float16* __restrict__ BPack,
                                              const int* __restrict__ src,
                                              const int* __restrict__ dst,
                                              const float* __restrict__ msgs,
                                              float* __restrict__ agg, int nTiles) {
  int lane = threadIdx.x & 31;
  int wid  = blockIdx.x * (blockDim.x >> 5) + (threadIdx.x >> 5);
  if (wid >= nTiles) return;
  int e0 = wid << 4;
  // stream-ahead hint for the next wave-tile of packed A (global_prefetch_b8)
  if (wid + 8 < nTiles)
    __builtin_prefetch(APack + (((size_t)(wid + 8) * 2) * 32 + lane) * 16, 0, 1);
  v8f c[8];
#pragma unroll
  for (int i = 0; i < 8; ++i) c[i] = zero8();
#pragma unroll
  for (int kt = 0; kt < 2; ++kt) {
    v16h a = load_pa(APack, wid, 2, kt, lane);
#pragma unroll
    for (int ct = 0; ct < 8; ++ct)
      c[ct] = wmma16(a, load_pb(BPack, kt, 8, ct, lane), c[ct]);
  }
  int mB = (lane >> 4) << 3, cl = lane & 15;
  int sA[8], dA[8];
#pragma unroll
  for (int v = 0; v < 8; ++v) {
    int e = e0 + mB + v;
    sA[v] = src[e];
    dA[v] = dst[e];
  }
#pragma unroll
  for (int ct = 0; ct < 8; ++ct)
#pragma unroll
    for (int v = 0; v < 8; ++v) {
      int col = ct * 16 + cl;
      float val = c[ct][v] + msgs[(size_t)sA[v] * 128 + col];
      atomicAdd(&agg[(size_t)dA[v] * 128 + col], val);
    }
}

// ==== act = has_msg ? hv@W_self + agg*inv_deg + b_msg : 0  (in_deg fixed => else==0) ====
__global__ __launch_bounds__(256) void k_act(const _Float16* __restrict__ APack,
                                             const _Float16* __restrict__ BPack,
                                             const float* __restrict__ agg,
                                             const float* __restrict__ indeg,
                                             const float* __restrict__ bmsg,
                                             float* __restrict__ act, int nTiles) {
  int lane = threadIdx.x & 31;
  int wid  = blockIdx.x * (blockDim.x >> 5) + (threadIdx.x >> 5);
  if (wid >= nTiles) return;
  int row0 = wid << 4;
  v8f c[8];
#pragma unroll
  for (int i = 0; i < 8; ++i) c[i] = zero8();
#pragma unroll
  for (int kt = 0; kt < 2; ++kt) {
    v16h a = load_pa(APack, wid, 2, kt, lane);
#pragma unroll
    for (int ct = 0; ct < 8; ++ct)
      c[ct] = wmma16(a, load_pb(BPack, kt, 8, ct, lane), c[ct]);
  }
  int mB = (lane >> 4) << 3, cl = lane & 15;
  float inv[8]; bool hm[8];
#pragma unroll
  for (int v = 0; v < 8; ++v) {
    float idg = indeg[row0 + mB + v];
    hm[v]  = idg > 0.0f;
    inv[v] = 1.0f / fmaxf(idg, 1.0f);
  }
#pragma unroll
  for (int ct = 0; ct < 8; ++ct)
#pragma unroll
    for (int v = 0; v < 8; ++v) {
      int col = ct * 16 + cl;
      size_t o = (size_t)(row0 + mB + v) * 128 + col;
      act[o] = hm[v] ? (c[ct][v] + agg[o] * inv[v] + bmsg[col]) : 0.0f;
    }
}

// ==== generic packed WMMA GEMM + bias ====
template <int KT, int CT>
__global__ __launch_bounds__(256) void k_gemm_bias(const _Float16* __restrict__ APack,
                                                   const _Float16* __restrict__ BPack,
                                                   const float* __restrict__ bias,
                                                   float* __restrict__ out, int ldo,
                                                   int nTiles) {
  int lane = threadIdx.x & 31;
  int wid  = blockIdx.x * (blockDim.x >> 5) + (threadIdx.x >> 5);
  if (wid >= nTiles) return;
  int row0 = wid << 4;
  v8f c[CT];
#pragma unroll
  for (int i = 0; i < CT; ++i) c[i] = zero8();
#pragma unroll
  for (int kt = 0; kt < KT; ++kt) {
    v16h a = load_pa(APack, wid, KT, kt, lane);
#pragma unroll
    for (int ct = 0; ct < CT; ++ct)
      c[ct] = wmma16(a, load_pb(BPack, kt, CT, ct, lane), c[ct]);
  }
  int mB = (lane >> 4) << 3, cl = lane & 15;
#pragma unroll
  for (int ct = 0; ct < CT; ++ct) {
    int col = ct * 16 + cl;
    float bv = bias[col];
#pragma unroll
    for (int v = 0; v < 8; ++v)
      out[(size_t)(row0 + mB + v) * ldo + col] = c[ct][v] + bv;
  }
}

// ==== GRU elementwise update (in-place hv) ====
__global__ void k_gru(const float* __restrict__ gi, const float* __restrict__ gh,
                      float* __restrict__ hv, int total) {
  int idx = blockIdx.x * blockDim.x + threadIdx.x;
  if (idx >= total) return;
  int n = idx >> 6, h = idx & 63;
  const float* gin = gi + (size_t)n * 192;
  const float* ghn = gh + (size_t)n * 192;
  float r  = sigf(gin[h] + ghn[h]);
  float z  = sigf(gin[64 + h] + ghn[64 + h]);
  float cc = tanhf(gin[128 + h] + r * ghn[128 + h]);
  hv[idx] = (1.0f - z) * cc + z * hv[idx];
}

// ==== x = sigmoid(hv@w_gate+b)*ns ; nd ====
__global__ void k_xns(const float* __restrict__ hv, const float* __restrict__ wg,
                      const float* __restrict__ bg, const float* __restrict__ indeg,
                      const float* __restrict__ outdeg, float* __restrict__ xs,
                      float* __restrict__ ndv, int N) {
  int n = blockIdx.x * blockDim.x + threadIdx.x;
  if (n >= N) return;
  float od = outdeg[n], idg = indeg[n];
  float ns = od > 0.0f ? rsqrtf(fmaxf(od, 1.0f)) : 0.0f;
  ndv[n]   = idg > 0.0f ? rsqrtf(fmaxf(idg, 1.0f)) : 0.0f;
  float acc[7];
#pragma unroll
  for (int j = 0; j < 7; ++j) acc[j] = bg[j];
  const float* h = hv + (size_t)n * 64;
  for (int k = 0; k < 64; ++k) {
    float v = h[k];
    const float* w = wg + k * 7;
#pragma unroll
    for (int j = 0; j < 7; ++j) acc[j] += v * w[j];
  }
#pragma unroll
  for (int j = 0; j < 7; ++j) xs[(size_t)n * 7 + j] = ns * sigf(acc[j]);
}

__global__ void k_convscatter(const int* __restrict__ src, const int* __restrict__ dst,
                              const float* __restrict__ xs, float* __restrict__ cacc, int E) {
  int e = blockIdx.x * blockDim.x + threadIdx.x;
  if (e >= E) return;
  int s = src[e], d = dst[e];
#pragma unroll
  for (int j = 0; j < 7; ++j) atomicAdd(&cacc[(size_t)d * 7 + j], xs[(size_t)s * 7 + j]);
}

// sum over nodes of nd[n]*(cacc[n]@w_conv)[col]; one block per col
__global__ __launch_bounds__(256) void k_graphreduce(const float* __restrict__ cacc,
                                                     const float* __restrict__ ndv,
                                                     const float* __restrict__ wconv,
                                                     float* __restrict__ gsum, int N) {
  int col = blockIdx.x;
  float wc[7];
#pragma unroll
  for (int j = 0; j < 7; ++j) wc[j] = wconv[j * 128 + col];
  float acc = 0.0f;
  for (int n = threadIdx.x; n < N; n += 256) {
    const float* cn = cacc + (size_t)n * 7;
    float s = 0.0f;
#pragma unroll
    for (int j = 0; j < 7; ++j) s += cn[j] * wc[j];
    acc += ndv[n] * s;
  }
  __shared__ float sh[256];
  sh[threadIdx.x] = acc; __syncthreads();
  for (int o = 128; o > 0; o >>= 1) {
    if (threadIdx.x < o) sh[threadIdx.x] += sh[threadIdx.x + o];
    __syncthreads();
  }
  if (threadIdx.x == 0) gsum[col] = sh[0];
}

// heads: graph_emb, base scalar, state_node/edge logits  (1 block, 128 threads)
__global__ __launch_bounds__(128) void k_head(const float* __restrict__ gsum,
                                              const float* __restrict__ bconv,
                                              const float* __restrict__ hv,
                                              const float* __restrict__ wch,
                                              const float* __restrict__ bch,
                                              const float* __restrict__ wsn,
                                              const float* __restrict__ bsn,
                                              const float* __restrict__ wse,
                                              const float* __restrict__ bse,
                                              float* __restrict__ out_ge,
                                              float* __restrict__ out_sn,
                                              float* __restrict__ out_se,
                                              float* __restrict__ scal, int N) {
  __shared__ float ge[128], ex[64], red[128];
  int t = threadIdx.x;
  ge[t] = gsum[t] * (1.0f / (float)N) + bconv[t];
  if (t < 64) ex[t] = hv[(size_t)(N - 1) * 64 + t];
  __syncthreads();
  float contrib = ge[t] * wch[t];
  if (t < 64) contrib += ex[t] * wch[128 + t];
  red[t] = contrib; __syncthreads();
  for (int o = 64; o > 0; o >>= 1) {
    if (t < o) red[t] += red[t + o];
    __syncthreads();
  }
  if (t == 0) scal[0] = red[0] + bch[0];
  out_ge[t] = ge[t];
  if (t < 3) {
    float sn = bsn[t], se = bse[t];
    for (int c2 = 0; c2 < 128; ++c2) { sn += ge[c2] * wsn[c2 * 3 + t]; se += ge[c2] * wse[c2 * 3 + t]; }
    for (int h2 = 0; h2 < 64; ++h2) se += ex[h2] * wse[(128 + h2) * 3 + t];
    out_sn[t] = sn; out_se[t] = se;
  }
}

__global__ __launch_bounds__(256) void k_logit(const float* __restrict__ hv,
                                               const float* __restrict__ wch,
                                               const float* __restrict__ scal,
                                               float* __restrict__ logits,
                                               float* __restrict__ pmax, int N) {
  int n = blockIdx.x * 256 + threadIdx.x;
  float l = -3.4e38f;
  if (n < N) {
    float acc = scal[0];
    const float* h = hv + (size_t)n * 64;
    for (int k = 0; k < 64; ++k) acc += h[k] * wch[192 + k];
    logits[n] = acc; l = acc;
  }
  __shared__ float sh[256];
  sh[threadIdx.x] = l; __syncthreads();
  for (int o = 128; o > 0; o >>= 1) {
    if (threadIdx.x < o) sh[threadIdx.x] = fmaxf(sh[threadIdx.x], sh[threadIdx.x + o]);
    __syncthreads();
  }
  if (threadIdx.x == 0) pmax[blockIdx.x] = sh[0];
}

__global__ __launch_bounds__(256) void k_redmax(const float* __restrict__ pmax,
                                                float* __restrict__ scal, int nb) {
  float acc = -3.4e38f;
  for (int i = threadIdx.x; i < nb; i += 256) acc = fmaxf(acc, pmax[i]);
  __shared__ float sh[256];
  sh[threadIdx.x] = acc; __syncthreads();
  for (int o = 128; o > 0; o >>= 1) {
    if (threadIdx.x < o) sh[threadIdx.x] = fmaxf(sh[threadIdx.x], sh[threadIdx.x + o]);
    __syncthreads();
  }
  if (threadIdx.x == 0) scal[1] = sh[0];
}

__global__ __launch_bounds__(256) void k_exp(const float* __restrict__ logits,
                                             const float* __restrict__ scal,
                                             float* __restrict__ evals,
                                             float* __restrict__ psum, int N) {
  int n = blockIdx.x * 256 + threadIdx.x;
  float e = 0.0f;
  if (n < N) { e = expf(logits[n] - scal[1]); evals[n] = e; }
  __shared__ float sh[256];
  sh[threadIdx.x] = e; __syncthreads();
  for (int o = 128; o > 0; o >>= 1) {
    if (threadIdx.x < o) sh[threadIdx.x] += sh[threadIdx.x + o];
    __syncthreads();
  }
  if (threadIdx.x == 0) psum[blockIdx.x] = sh[0];
}

__global__ __launch_bounds__(256) void k_redsum(const float* __restrict__ psum,
                                                float* __restrict__ scal, int nb) {
  float acc = 0.0f;
  for (int i = threadIdx.x; i < nb; i += 256) acc += psum[i];
  __shared__ float sh[256];
  sh[threadIdx.x] = acc; __syncthreads();
  for (int o = 128; o > 0; o >>= 1) {
    if (threadIdx.x < o) sh[threadIdx.x] += sh[threadIdx.x + o];
    __syncthreads();
  }
  if (threadIdx.x == 0) scal[2] = sh[0];
}

__global__ void k_norm(const float* __restrict__ evals, const float* __restrict__ scal,
                       float* __restrict__ out, int N) {
  int n = blockIdx.x * blockDim.x + threadIdx.x;
  if (n < N) out[n] = evals[n] / scal[2];
}

// ================= host =================
static inline void zero_async(float* p, size_t n, hipStream_t s) {
  int blocks = (int)((n + 255) / 256);
  if (blocks > 4096) blocks = 4096;
  k_zero<<<blocks, 256, 0, s>>>(p, n);
}

static inline void pack_a_async(const float* A, int lda, int KT, int nRowTiles,
                                _Float16* out, hipStream_t s) {
  size_t total = (size_t)nRowTiles * KT * 512;
  int blocks = (int)((total + 255) / 256);
  if (blocks > 16384) blocks = 16384;
  k_pack_a<<<blocks, 256, 0, s>>>(A, lda, KT, nRowTiles, out);
}

static inline void pack_b_async(const float* B, int ldb, int KTt, int CTt,
                                _Float16* out, hipStream_t s) {
  size_t total = (size_t)KTt * CTt * 512;
  int blocks = (int)((total + 255) / 256);
  k_pack_b<<<blocks, 256, 0, s>>>(B, ldb, KTt, CTt, out);
}

extern "C" void kernel_launch(void* const* d_in, const int* in_sizes, int n_in,
                              void* d_out, int out_size, void* d_ws, size_t ws_size,
                              hipStream_t stream) {
  const float* hv_in = (const float*)d_in[0];
  const float* he    = (const float*)d_in[1];
  const int*   src   = (const int*)d_in[2];
  const int*   dst   = (const int*)d_in[3];
  const float* Wself = (const float*)d_in[4];
  const float* Wsrc  = (const float*)d_in[5];
  const float* Wedge = (const float*)d_in[6];
  const float* bmsg  = (const float*)d_in[7];
  const float* wih   = (const float*)d_in[8];
  const float* whh   = (const float*)d_in[9];
  const float* bih   = (const float*)d_in[10];
  const float* bhh   = (const float*)d_in[11];
  const float* wgate = (const float*)d_in[12];
  const float* bgate = (const float*)d_in[13];
  const float* wconv = (const float*)d_in[14];
  const float* bconv = (const float*)d_in[15];
  const float* wch   = (const float*)d_in[16];
  const float* bch   = (const float*)d_in[17];
  const float* wsn   = (const float*)d_in[18];
  const float* bsn   = (const float*)d_in[19];
  const float* wse   = (const float*)d_in[20];
  const float* bse   = (const float*)d_in[21];

  const int N = in_sizes[0] / 64;
  const int E = in_sizes[2];
  const int R = in_sizes[7] / 128;

  // ---- fp32 workspace partition ----
  float* F = (float*)d_ws;
  size_t o = 0;
  float* hv    = F + o; o += (size_t)N * 64;
  float* msgs  = F + o; o += (size_t)N * 128;   // reused as `act` after k_edge
  float* agg   = F + o; o += (size_t)N * 128;
  float* gi    = F + o; o += (size_t)N * 192;
  float* gh    = F + o; o += (size_t)N * 192;
  float* indeg = F + o; o += (size_t)N;
  float* outdeg= F + o; o += (size_t)N;
  float* ndv   = F + o; o += (size_t)N;
  float* xs    = F + o; o += (size_t)N * 7;
  float* cacc  = F + o; o += (size_t)N * 7;
  float* gsum  = F + o; o += 128;
  float* logits= F + o; o += (size_t)N;
  float* evals = F + o; o += (size_t)N;
  float* pmax  = F + o; o += 1024;
  float* psum  = F + o; o += 1024;
  float* scal  = F + o; o += 8;
  float* act   = msgs;  // alias: act overwrites msgs (dead after edge kernel)

  // ---- f16 packed-fragment region (64B aligned) ----
  size_t byteOff = (o * sizeof(float) + 63) & ~(size_t)63;
  _Float16* Hb = (_Float16*)((char*)d_ws + byteOff);
  size_t ho = 0;
  _Float16* hePack   = Hb + ho; ho += (size_t)E * 64;    // A frags, KT=2
  _Float16* hvPack   = Hb + ho; ho += (size_t)N * 64;    // A frags, KT=2
  _Float16* actPack  = Hb + ho; ho += (size_t)N * 128;   // A frags, KT=4
  _Float16* wsrcPack = Hb + ho; ho += 2 * 8 * 512;
  _Float16* wedgePack= Hb + ho; ho += 2 * 8 * 512;
  _Float16* wselfPack= Hb + ho; ho += 2 * 8 * 512;
  _Float16* wihPack  = Hb + ho; ho += 4 * 12 * 512;
  _Float16* whhPack  = Hb + ho; ho += 2 * 12 * 512;
  (void)ws_size; (void)n_in; (void)out_size;

  float* out        = (float*)d_out;
  float* out_hv     = out;
  float* out_ge     = out + (size_t)N * 64;
  float* out_choose = out_ge + 128;
  float* out_sn     = out_choose + N;
  float* out_se     = out_sn + 3;

  const int nNodeTiles = N / 16;
  const int nEdgeTiles = E / 16;
  const int nodeGemmBlks = (nNodeTiles + 7) / 8;
  const int edgeGemmBlks = (nEdgeTiles + 7) / 8;

  hipMemcpyAsync(hv, hv_in, (size_t)N * 64 * sizeof(float), hipMemcpyDeviceToDevice, stream);
  zero_async(indeg, N, stream);
  zero_async(outdeg, N, stream);
  zero_async(cacc, (size_t)N * 7, stream);
  k_deg<<<(E + 255) / 256, 256, 0, stream>>>(src, dst, indeg, outdeg, E);
  pack_a_async(he, 64, 2, nEdgeTiles, hePack, stream);   // once; reused both rounds

  for (int t = 0; t < R; ++t) {
    // pack per-round operands into fragment order
    pack_a_async(hv, 64, 2, nNodeTiles, hvPack, stream);
    pack_b_async(Wsrc  + (size_t)t * 64 * 128, 128, 2, 8,  wsrcPack,  stream);
    pack_b_async(Wedge + (size_t)t * 64 * 128, 128, 2, 8,  wedgePack, stream);
    pack_b_async(Wself + (size_t)t * 64 * 128, 128, 2, 8,  wselfPack, stream);
    pack_b_async(wih   + (size_t)t * 128 * 192, 192, 4, 12, wihPack,  stream);
    pack_b_async(whh   + (size_t)t * 64 * 192, 192, 2, 12, whhPack,   stream);

    zero_async(agg, (size_t)N * 128, stream);
    k_msgsrc<<<nodeGemmBlks, 256, 0, stream>>>(hvPack, wsrcPack, msgs, nNodeTiles);
    k_edge<<<edgeGemmBlks, 256, 0, stream>>>(hePack, wedgePack, src, dst, msgs, agg, nEdgeTiles);
    k_act<<<nodeGemmBlks, 256, 0, stream>>>(hvPack, wselfPack, agg, indeg,
                                            bmsg + (size_t)t * 128, act, nNodeTiles);
    pack_a_async(act, 128, 4, nNodeTiles, actPack, stream);
    k_gemm_bias<4, 12><<<nodeGemmBlks, 256, 0, stream>>>(actPack, wihPack,
                                                         bih + (size_t)t * 192, gi, 192, nNodeTiles);
    k_gemm_bias<2, 12><<<nodeGemmBlks, 256, 0, stream>>>(hvPack, whhPack,
                                                         bhh + (size_t)t * 192, gh, 192, nNodeTiles);
    k_gru<<<((size_t)N * 64 + 255) / 256, 256, 0, stream>>>(gi, gh, hv, N * 64);
  }

  k_xns<<<(N + 255) / 256, 256, 0, stream>>>(hv, wgate, bgate, indeg, outdeg, xs, ndv, N);
  k_convscatter<<<(E + 255) / 256, 256, 0, stream>>>(src, dst, xs, cacc, E);
  k_graphreduce<<<128, 256, 0, stream>>>(cacc, ndv, wconv, gsum, N);
  k_head<<<1, 128, 0, stream>>>(gsum, bconv, hv, wch, bch, wsn, bsn, wse, bse,
                                out_ge, out_sn, out_se, scal, N);

  hipMemcpyAsync(out_hv, hv, (size_t)N * 64 * sizeof(float), hipMemcpyDeviceToDevice, stream);

  const int nb = (N + 255) / 256;
  k_logit<<<nb, 256, 0, stream>>>(hv, wch, scal, logits, pmax, N);
  k_redmax<<<1, 256, 0, stream>>>(pmax, scal, nb);
  k_exp<<<nb, 256, 0, stream>>>(logits, scal, evals, psum, N);
  k_redsum<<<1, 256, 0, stream>>>(psum, scal, nb);
  k_norm<<<(N + 255) / 256, 256, 0, stream>>>(evals, scal, out_choose, N);
}
